// LocationSensitiveAttention_44787918962877
// MI455X (gfx1250) — matrix operand order repaired and checked by reading, not compiled, not run on hardware
//
#include <hip/hip_runtime.h>
#include <hip/hip_bf16.h>

typedef __attribute__((ext_vector_type(16))) __bf16 v16bf;
typedef __attribute__((ext_vector_type(8)))  float  v8f;
typedef __attribute__((ext_vector_type(4)))  float  v4f;

#define B_    64
#define T_    1000
#define DEC_  1024
#define ENC_  512
#define ATT_  128
#define FILT_ 32
#define KS_   31
#define PAD_  15

// ---- helpers -------------------------------------------------------------

__device__ inline void loadf8(const float* __restrict__ p, float* d) {
  v4f x = *(const v4f*)(p);
  v4f y = *(const v4f*)(p + 4);
#pragma unroll
  for (int i = 0; i < 4; ++i) { d[i] = x[i]; d[4 + i] = y[i]; }
}

// A-tile fragment (16x32 bf16): elements 0..7 <- K kbase1..+7, 8..15 <- kbase2..+7
__device__ inline v16bf packA(const float* lo, const float* hi) {
  v16bf r;
#pragma unroll
  for (int i = 0; i < 8; ++i) { r[i] = (__bf16)lo[i]; r[8 + i] = (__bf16)hi[i]; }
  return r;
}

// B-tile fragment (32x16 bf16): elements 0..7 <- K kbase1..+7, 8..15 <- kbase2..+7
__device__ inline v16bf packB(const float* p) {
  v16bf r;
#pragma unroll
  for (int i = 0; i < 16; ++i) r[i] = (__bf16)p[i];
  return r;
}

// ---- kernel 1: query = decoder_hidden @ query_w^T  (64x128, K=1024) ------

__global__ __launch_bounds__(32)
void query_gemm_kernel(const float* __restrict__ dh,
                       const float* __restrict__ qw,
                       float* __restrict__ query) {
  const int mtile = blockIdx.x;     // 0..3
  const int atile = blockIdx.y;     // 0..7
  const int lane  = threadIdx.x;
  const int n     = lane & 15;
  const int hi    = (lane >> 4) & 1;
  const int m_a   = mtile * 16 + n;           // A row for this lane
  const int acol  = atile * 16 + n;           // B column for this lane
  const float* arow = dh + (size_t)m_a * DEC_;
  const float* brow = qw + (size_t)acol * DEC_;
  const int ka = hi ? 8 : 0;                  // A lane K sub-offset
  const int kb = hi ? 8 : 0;                  // B lane K sub-offset

  v8f c = {};
  for (int k0 = 0; k0 < DEC_; k0 += 32) {
    float ta[16], tb[16];
    loadf8(arow + k0 + ka, ta);
    loadf8(arow + k0 + ka + 16, ta + 8);
    loadf8(brow + k0 + kb, tb);
    loadf8(brow + k0 + kb + 16, tb + 8);
    v16bf av = packA(ta, ta + 8);
    v16bf bv = packB(tb);
    c = __builtin_amdgcn_wmma_f32_16x16x32_bf16(false, av, false, bv,
                                                (short)0, c, false, false);
  }
  const int mbase = hi * 8;
#pragma unroll
  for (int i = 0; i < 8; ++i) {
    int m = mtile * 16 + mbase + i;
    query[(size_t)m * ATT_ + acol] = c[i];
  }
}

// ---- kernel 2: fused conv + loc-proj + keys GEMM + tanh + value dot ------
// grid: (B, ceil(T/16)), block: 256 (8 waves; wave w owns ATT tile w)

__global__ __launch_bounds__(256)
void fused_energies_kernel(const float* __restrict__ enc,
                           const float* __restrict__ pw,
                           const float* __restrict__ pwc,
                           const float* __restrict__ conv_w,
                           const float* __restrict__ loc_proj_w,
                           const float* __restrict__ key_w,
                           const float* __restrict__ value_w,
                           const float* __restrict__ query,
                           float* __restrict__ energies) {
  __shared__ float s_pw[48];
  __shared__ float s_pwc[48];
  __shared__ float s_conv[16 * 32];   // conv output tile [t_local][filter]
  __shared__ float s_energy[16];

  const int b   = blockIdx.x;
  const int t0  = blockIdx.y * 16;
  const int tid = threadIdx.x;

  if (tid < 16) s_energy[tid] = 0.0f;

  // stage prev_weights / prev_weights_cum window (16 + 2*PAD = 46 values)
  if (tid < 46) {
    int tg = t0 + tid - PAD_;
    s_pw[tid] = (tg >= 0 && tg < T_) ? pw[(size_t)b * T_ + tg] : 0.0f;
  } else if (tid >= 64 && tid < 110) {
    int j  = tid - 64;
    int tg = t0 + j - PAD_;
    s_pwc[j] = (tg >= 0 && tg < T_) ? pwc[(size_t)b * T_ + tg] : 0.0f;
  }
  __syncthreads();

  // conv1d: 16 t x 32 filters = 512 outputs, 2 per thread
  for (int idx = tid; idx < 512; idx += 256) {
    int m = idx >> 5;     // t_local
    int f = idx & 31;
    float acc = 0.0f;
#pragma unroll
    for (int k = 0; k < KS_; ++k) {
      acc += s_pw[m + k]  * conv_w[(f * 2 + 0) * KS_ + k];
      acc += s_pwc[m + k] * conv_w[(f * 2 + 1) * KS_ + k];
    }
    s_conv[m * 32 + f] = acc;
  }
  __syncthreads();

  // per-wave WMMA: keys tile (16t x 16a, K=512) + location projection (K=32)
  const int wave  = tid >> 5;
  const int lane  = tid & 31;
  const int n     = lane & 15;
  const int hi    = (lane >> 4) & 1;
  const int acol  = wave * 16 + n;
  const int mbase = hi * 8;
  const int ka    = hi ? 8 : 0;
  const int kb    = hi ? 8 : 0;

  const int   t_a    = t0 + n;            // A row (t) this lane supplies
  const bool  tvalid = (t_a < T_);
  const float* arow  = enc + ((size_t)b * T_ + (size_t)t_a) * ENC_;
  const float* brow  = key_w + (size_t)acol * ENC_;

  v8f c = {};
  for (int k0 = 0; k0 < ENC_; k0 += 32) {
    float ta[16], tb[16];
    if (tvalid) {
      loadf8(arow + k0 + ka, ta);
      loadf8(arow + k0 + ka + 16, ta + 8);
    } else {
#pragma unroll
      for (int i = 0; i < 16; ++i) ta[i] = 0.0f;
    }
    loadf8(brow + k0 + kb, tb);
    loadf8(brow + k0 + kb + 16, tb + 8);
    v16bf av = packA(ta, ta + 8);
    v16bf bv = packB(tb);
    c = __builtin_amdgcn_wmma_f32_16x16x32_bf16(false, av, false, bv,
                                                (short)0, c, false, false);
  }

  // location projection: A = conv tile (16x32), B = loc_proj_w[a][f]
  {
    float ta[16], tb[16];
#pragma unroll
    for (int i = 0; i < 8; ++i) {
      ta[i]     = s_conv[n * 32 + ka + i];
      ta[8 + i] = s_conv[n * 32 + ka + 16 + i];
    }
    loadf8(loc_proj_w + (size_t)acol * FILT_ + kb, tb);
    loadf8(loc_proj_w + (size_t)acol * FILT_ + kb + 16, tb + 8);
    v16bf av = packA(ta, ta + 8);
    v16bf bv = packB(tb);
    c = __builtin_amdgcn_wmma_f32_16x16x32_bf16(false, av, false, bv,
                                                (short)0, c, false, false);
  }

  // epilogue: + query, tanh, * value_w, reduce over ATT into per-t energy
  const float qv = query[(size_t)b * ATT_ + acol];
  const float vw = value_w[acol];
#pragma unroll
  for (int i = 0; i < 8; ++i) {
    int m = mbase + i;
    if (t0 + m < T_) {
      float h = tanhf(c[i] + qv);
      atomicAdd(&s_energy[m], h * vw);
    }
  }
  __syncthreads();

  if (tid < 16) {
    int t = t0 + tid;
    if (t < T_) energies[(size_t)b * T_ + t] = s_energy[tid];
  }
}

// ---- kernel 3: masked softmax over T + context = attn @ enc --------------

__global__ __launch_bounds__(256)
void softmax_context_kernel(const float* __restrict__ energies,
                            const unsigned char* __restrict__ mask,
                            const float* __restrict__ enc,
                            float* __restrict__ out) {
  __shared__ float s_e[T_];
  __shared__ float red[256];

  const int b   = blockIdx.x;
  const int tid = threadIdx.x;

  float lmax = -INFINITY;
  for (int t = tid; t < T_; t += 256) {
    float e = energies[(size_t)b * T_ + t];
    if (mask[(size_t)b * T_ + t]) e = -INFINITY;
    s_e[t] = e;
    lmax = fmaxf(lmax, e);
  }
  red[tid] = lmax;
  __syncthreads();
  for (int s = 128; s > 0; s >>= 1) {
    if (tid < s) red[tid] = fmaxf(red[tid], red[tid + s]);
    __syncthreads();
  }
  const float gmax = red[0];
  __syncthreads();

  float lsum = 0.0f;
  for (int t = tid; t < T_; t += 256) {
    float w = __expf(s_e[t] - gmax);
    s_e[t] = w;
    lsum += w;
  }
  red[tid] = lsum;
  __syncthreads();
  for (int s = 128; s > 0; s >>= 1) {
    if (tid < s) red[tid] += red[tid + s];
    __syncthreads();
  }
  const float inv = 1.0f / red[0];
  __syncthreads();

  float* attn_out = out + (size_t)B_ * ENC_;
  for (int t = tid; t < T_; t += 256) {
    float w = s_e[t] * inv;
    s_e[t] = w;
    attn_out[(size_t)b * T_ + t] = w;
  }
  __syncthreads();

  // context: each thread owns 2 encoder columns (512 / 256)
  const float* encb = enc + (size_t)b * T_ * ENC_;
  const int e0 = tid, e1 = tid + 256;
  float acc0 = 0.0f, acc1 = 0.0f;
  for (int t = 0; t < T_; ++t) {
    float w = s_e[t];
    acc0 += w * encb[(size_t)t * ENC_ + e0];
    acc1 += w * encb[(size_t)t * ENC_ + e1];
  }
  out[(size_t)b * ENC_ + e0] = acc0;
  out[(size_t)b * ENC_ + e1] = acc1;
}

// ---- launcher ------------------------------------------------------------

extern "C" void kernel_launch(void* const* d_in, const int* in_sizes, int n_in,
                              void* d_out, int out_size, void* d_ws, size_t ws_size,
                              hipStream_t stream) {
  (void)in_sizes; (void)n_in; (void)out_size; (void)ws_size;

  const float* dh          = (const float*)d_in[0];  // (B, DEC)
  const float* enc         = (const float*)d_in[1];  // (B, T, ENC)
  const float* pw          = (const float*)d_in[2];  // (B, T)
  const float* pwc         = (const float*)d_in[3];  // (B, T)
  const unsigned char* msk = (const unsigned char*)d_in[4]; // (B, T) bool
  const float* conv_w      = (const float*)d_in[5];  // (FILT, 2, KS)
  const float* loc_proj_w  = (const float*)d_in[6];  // (ATT, FILT)
  const float* query_w     = (const float*)d_in[7];  // (ATT, DEC)
  const float* key_w       = (const float*)d_in[8];  // (ATT, ENC)
  const float* value_w     = (const float*)d_in[9];  // (1, ATT)

  float* out      = (float*)d_out;                   // context | attn
  float* ws       = (float*)d_ws;
  float* query    = ws;                              // B*ATT floats
  float* energies = ws + (size_t)B_ * ATT_;          // B*T floats

  query_gemm_kernel<<<dim3(4, 8), 32, 0, stream>>>(dh, query_w, query);

  fused_energies_kernel<<<dim3(B_, (T_ + 15) / 16), 256, 0, stream>>>(
      enc, pw, pwc, conv_w, loc_proj_w, key_w, value_w, query, energies);

  softmax_context_kernel<<<B_, 256, 0, stream>>>(energies, msk, enc, out);
}
